// SwinTransformer_29205777612880
// MI455X (gfx1250) — matrix operand order repaired and checked by reading, not compiled, not run on hardware
//
#include <hip/hip_runtime.h>
#include <hip/hip_bf16.h>
#include <math.h>

typedef __attribute__((ext_vector_type(16))) _Float16 v16h;
typedef __attribute__((ext_vector_type(8)))  float    v8f;
typedef __attribute__((ext_vector_type(4)))  unsigned int u32x4;
typedef __attribute__((ext_vector_type(8)))  int      i32x8;
typedef __attribute__((ext_vector_type(4)))  int      i32x4;

// ---- WMMA fragment K-index patterns (CDNA5 ISA 7.12.2, wave32) ----
// A-matrix 16x32 f16: VGPR v holds K = (v/4)*16 + (v%4)*2 (+8 if lane>=16), pair (K,K+1)
__device__ __forceinline__ int kpatA(int v, int g) { return ((v >> 2) << 4) + ((v & 3) << 1) + (g << 3); }
// B-matrix 32x16 f16: VGPR v holds K = (v/4)*16 + (v%4)*2 (+8 if lane>=16), pair (K,K+1)
__device__ __forceinline__ int kpatB(int v, int g) { return ((v >> 2) << 4) + ((v & 3) << 1) + (g << 3); }

// ============================================================================
// TDM: async 2D tile load global->LDS (CDNA5 ISA ch.8, D# groups per 8.3-8.6).
// Tile: tileRows x 32 fp32 elements, row stride = strideElems (elements).
// 2D tensor: groups 2/3 zero (tile_dim2/3 = 0 => unused); not in a cluster so
// workgroup_mask = 0. Tracked by TENSORcnt.
// This toolchain exposes the 6-arg builtin form (extra i32x8 group, zeroed).
// ============================================================================
__device__ __forceinline__ void tdm_load_tile(const float* src, unsigned ldsByteOff,
                                              int tileRows, int strideElems)
{
  unsigned long long ga = (unsigned long long)(size_t)src;
  u32x4 g0;
  g0[0] = 1u;                                                  // count=1 (valid), user mode
  g0[1] = ldsByteOff;                                          // lds_addr (bytes)
  g0[2] = (unsigned)(ga & 0xffffffffull);                      // global_addr[31:0]
  g0[3] = (unsigned)((ga >> 32) & 0x1ffffffull) | (2u << 30);  // global_addr[56:32] | type=2
  i32x8 g1;
  g1[0] = (int)(2u << 16);                                     // wg_mask=0, data_size=4B
  g1[1] = (int)(32u << 16);                                    // tensor_dim0 = 32 (bits 63:48)
  g1[2] = (int)(((unsigned)tileRows & 0xffffu) << 16);         // tensor_dim1[15:0] (bits 95:80)
  g1[3] = (int)(32u << 16);                                    // tile_dim0 = 32 (bits 127:112)
  g1[4] = (int)((unsigned)tileRows & 0xffffu);                 // tile_dim1 (bits 143:128)
  g1[5] = strideElems;                                         // tensor_dim0_stride[31:0]
  g1[6] = 0;                                                   // stride[47:32]=0, dim1_stride lo=0
  g1[7] = 0;                                                   // dim1_stride hi = 0
  i32x4 gz = {0, 0, 0, 0};                                     // groups 2/3: 2D -> unused
  i32x8 gz8 = {0, 0, 0, 0, 0, 0, 0, 0};                        // extra group (6-arg form)
  __builtin_amdgcn_tensor_load_to_lds(g0, g1, gz, gz, gz8, 0);
}

// ============================================================================
// Generic GEMM: Out[M,N] = act(A[M,K] @ W[N,K]^T + bias[N])
// 256 threads = 8 waves; block tile 64(M) x 32(N); each wave one 16x16 tile.
// Double-buffered fp32 LDS staging via TDM (wave 0 issues, s_wait_tensorcnt),
// f32->f16 conversion during fragment build, 16x16x32 f16 WMMA accumulate.
// M multiple of 64, N and K multiples of 32 (guaranteed by network shapes).
// ============================================================================
template <int ACT> // 0 = none, 1 = exact GELU
__global__ __launch_bounds__(256) void gemm_wmma_kernel(
    const float* __restrict__ A, const float* __restrict__ Wt,
    const float* __restrict__ bias, float* __restrict__ Out,
    int N, int K)
{
  // Single shared object => struct-layout offsets are the LDS byte offsets.
  __shared__ struct { float A[2][64][32]; float B[2][32][32]; } sm;
  const unsigned LDS_A = 0u;
  const unsigned LDS_B = (unsigned)(2 * 64 * 32 * sizeof(float));

  const int t = threadIdx.x;
  const int lane = t & 31, wv = t >> 5;
  const int g = lane >> 4, ln_ = lane & 15;
  const int blockM = blockIdx.y * 64;
  const int blockN = blockIdx.x * 32;
  const int mt = (wv >> 1) * 16; // 0,16,32,48
  const int nt = (wv & 1) * 16;  // 0,16

  const float* Abase = A  + (size_t)blockM * K;
  const float* Bbase = Wt + (size_t)blockN * K;

  if (wv == 0) {                         // wave-uniform: TDM issues once per wave
    tdm_load_tile(Abase, LDS_A, 64, K);
    tdm_load_tile(Bbase, LDS_B, 32, K);
    __builtin_amdgcn_s_wait_tensorcnt(0);
  }
  __syncthreads();

  v8f acc = {};
  int buf = 0;
  for (int k0 = 0; k0 < K; k0 += 32) {
    const int nxt = buf ^ 1;
    const bool more = (k0 + 32) < K;
    if (more && wv == 0) {
      tdm_load_tile(Abase + k0 + 32, LDS_A + (unsigned)nxt * (64 * 32 * 4), 64, K);
      tdm_load_tile(Bbase + k0 + 32, LDS_B + (unsigned)nxt * (32 * 32 * 4), 32, K);
    }
    v16h a, b;
#pragma unroll
    for (int v = 0; v < 8; ++v) {
      int ka = kpatA(v, g);
      a[2 * v]     = (_Float16)sm.A[buf][mt + ln_][ka];
      a[2 * v + 1] = (_Float16)sm.A[buf][mt + ln_][ka + 1];
      int kb = kpatB(v, g);
      b[2 * v]     = (_Float16)sm.B[buf][nt + ln_][kb];
      b[2 * v + 1] = (_Float16)sm.B[buf][nt + ln_][kb + 1];
    }
    acc = __builtin_amdgcn_wmma_f32_16x16x32_f16(false, a, false, b, (short)0, acc, false, false);
    if (more && wv == 0) __builtin_amdgcn_s_wait_tensorcnt(0);
    __syncthreads();
    buf = nxt;
  }
#pragma unroll
  for (int r = 0; r < 8; ++r) {
    int m = blockM + mt + r + 8 * g;
    int n = blockN + nt + ln_;
    float val = acc[r] + (bias ? bias[n] : 0.0f);
    if (ACT == 1) val = 0.5f * val * (1.0f + erff(val * 0.70710678118654752f));
    Out[(size_t)m * N + n] = val;
  }
}

// ============================================================================
// Patch embed: 4x4/stride-4 conv, 3->96, NCHW input -> NHWC tokens
// ============================================================================
__global__ void patch_embed_kernel(const float* __restrict__ x,
                                   const float* __restrict__ w,
                                   const float* __restrict__ bias,
                                   float* __restrict__ out)
{
  size_t idx = (size_t)blockIdx.x * 256 + threadIdx.x;
  const size_t total = (size_t)16 * 64 * 64 * 96;
  if (idx >= total) return;
  int oc = idx % 96;
  size_t p = idx / 96;
  int ow = p % 64, oh = (p / 64) % 64, b = p / (64 * 64);
  float acc = bias[oc];
#pragma unroll
  for (int ci = 0; ci < 3; ++ci)
#pragma unroll
    for (int kh = 0; kh < 4; ++kh)
#pragma unroll
      for (int kw = 0; kw < 4; ++kw)
        acc += x[((size_t)(b * 3 + ci) * 256 + oh * 4 + kh) * 256 + ow * 4 + kw] *
               w[((oc * 3 + ci) * 4 + kh) * 4 + kw];
  out[idx] = acc;
}

// ============================================================================
// LayerNorm (wave per token). out = (resid ? resid + LN(y) : LN(y))
// ============================================================================
__global__ __launch_bounds__(256) void ln_kernel(
    const float* __restrict__ y, const float* __restrict__ g,
    const float* __restrict__ b, const float* __restrict__ resid,
    float* __restrict__ out, int C)
{
  int token = blockIdx.x * 8 + (threadIdx.x >> 5);
  int lane = threadIdx.x & 31;
  const float* yr = y + (size_t)token * C;
  float s = 0.f, ss = 0.f;
  for (int c = lane; c < C; c += 32) { float v = yr[c]; s += v; ss += v * v; }
#pragma unroll
  for (int o = 16; o; o >>= 1) { s += __shfl_xor(s, o, 32); ss += __shfl_xor(ss, o, 32); }
  float mean = s / C;
  float var = ss / C - mean * mean;
  float inv = rsqrtf(fmaxf(var, 0.0f) + 1e-5f);
  float* orow = out + (size_t)token * C;
  const float* rr = resid ? resid + (size_t)token * C : nullptr;
  for (int c = lane; c < C; c += 32) {
    float v = (yr[c] - mean) * inv * g[c] + b[c];
    orow[c] = rr ? rr[c] + v : v;
  }
}

// ============================================================================
// Window partition with cyclic shift (roll by -sh): x(B,H,W,C) -> xw(nWin,64,C)
// ============================================================================
__global__ void win_part_kernel(const float* __restrict__ x, float* __restrict__ xw,
                                int Hc, int Cc, int sh, int nW)
{
  size_t idx = (size_t)blockIdx.x * 256 + threadIdx.x;
  size_t total = (size_t)16 * Hc * Hc * Cc;
  if (idx >= total) return;
  int c = idx % Cc;
  size_t p = idx / Cc;
  int n = p % 64;
  size_t wIdx = p / 64;
  int ww = wIdx % nW;
  size_t t = wIdx / nW;
  int wh = t % nW, b = t / nW;
  int hh = (wh * 8 + (n >> 3) + sh) % Hc;
  int wc = (ww * 8 + (n & 7) + sh) % Hc;
  xw[idx] = x[(((size_t)b * Hc + hh) * Hc + wc) * Cc + c];
}

// Window reverse with roll-back (+sh): xw(nWin,64,C) -> x(B,H,W,C)
__global__ void win_rev_kernel(const float* __restrict__ xw, float* __restrict__ x,
                               int Hc, int Cc, int sh, int nW)
{
  size_t idx = (size_t)blockIdx.x * 256 + threadIdx.x;
  size_t total = (size_t)16 * Hc * Hc * Cc;
  if (idx >= total) return;
  int c = idx % Cc;
  size_t p = idx / Cc;
  int w = p % Hc;
  int h = (p / Hc) % Hc;
  int b = p / ((size_t)Hc * Hc);
  int hr = (h - sh + Hc) % Hc;
  int wr = (w - sh + Hc) % Hc;
  size_t wIdx = ((size_t)b * nW + (hr >> 3)) * nW + (wr >> 3);
  int n = ((hr & 7) << 3) | (wr & 7);
  x[idx] = xw[(wIdx * 64 + n) * Cc + c];
}

// ============================================================================
// Patch-merge gather: (B,H,W,C) -> (B,H/2,W/2,4C), Swin concat order
// ============================================================================
__global__ void merge_gather_kernel(const float* __restrict__ x, float* __restrict__ out,
                                    int Hc, int Cc)
{
  int H2 = Hc / 2;
  size_t idx = (size_t)blockIdx.x * 256 + threadIdx.x;
  size_t total = (size_t)16 * H2 * H2 * 4 * Cc;
  if (idx >= total) return;
  int cc = idx % (4 * Cc);
  size_t p = idx / (4 * Cc);
  int j = p % H2;
  int i = (p / H2) % H2;
  int b = p / ((size_t)H2 * H2);
  int blk = cc / Cc, c = cc % Cc;
  int hs = 2 * i + (blk & 1);
  int ws = 2 * j + (blk >> 1);
  out[idx] = x[(((size_t)b * Hc + hs) * Hc + ws) * Cc + c];
}

// ============================================================================
// Continuous-position-bias MLP: 15x15 log-spaced rel coords -> 512 relu -> heads,
// then 16*sigmoid, scattered to biasTab[head][64][64]. One block.
// ============================================================================
__global__ __launch_bounds__(256) void cpb_kernel(
    const float* __restrict__ w1, const float* __restrict__ b1,
    const float* __restrict__ w2, float* __restrict__ biasTab, int heads)
{
  __shared__ float cpbS[225 * 24];
  int t = threadIdx.x;
  for (int r = t; r < 225; r += 256) {
    float di = (float)(r / 15 - 7), dj = (float)(r % 15 - 7);
    float t0 = di * (8.0f / 7.0f), t1 = dj * (8.0f / 7.0f);
    t0 = (t0 > 0.f ? 1.f : (t0 < 0.f ? -1.f : 0.f)) * log2f(fabsf(t0) + 1.f) * (1.0f / 3.0f);
    t1 = (t1 > 0.f ? 1.f : (t1 < 0.f ? -1.f : 0.f)) * log2f(fabsf(t1) + 1.f) * (1.0f / 3.0f);
    float acc[24];
    for (int h = 0; h < heads; ++h) acc[h] = 0.f;
    for (int k = 0; k < 512; ++k) {
      float hv = fmaxf(w1[2 * k] * t0 + w1[2 * k + 1] * t1 + b1[k], 0.f);
      for (int h = 0; h < heads; ++h) acc[h] += w2[h * 512 + k] * hv;
    }
    for (int h = 0; h < heads; ++h)
      cpbS[r * heads + h] = 16.f / (1.f + expf(-acc[h]));
  }
  __syncthreads();
  for (int pos = t; pos < 64 * 64; pos += 256) {
    int m = pos >> 6, n = pos & 63;
    int ridx = ((m >> 3) - (n >> 3) + 7) * 15 + ((m & 7) - (n & 7) + 7);
    for (int h = 0; h < heads; ++h)
      biasTab[h * 4096 + pos] = cpbS[ridx * heads + h];
  }
}

// ============================================================================
// Fused Swin-V2 window attention, one block per (window, head), head_dim=32.
// 128 threads = 4 waves. Cosine attention + CPB bias + optional shift mask,
// S = Qn Kn^T via 16 WMMAs, softmax fp32, O = P V via 16 WMMAs (8 chains of 2).
// ============================================================================
__global__ __launch_bounds__(128) void attn_kernel(
    const float* __restrict__ q, const float* __restrict__ k,
    const float* __restrict__ v, const float* __restrict__ lsc,
    const float* __restrict__ biasTab, float* __restrict__ o,
    int Cc, int nW, int Hc, int sh)
{
  __shared__ _Float16 Qh[64][32];
  __shared__ _Float16 Kh[64][32];
  __shared__ _Float16 Vh[64][32];
  __shared__ float    S[64][64];
  __shared__ _Float16 P[64][64];

  const int win = blockIdx.x, head = blockIdx.y;
  const int t = threadIdx.x;
  const size_t base = (size_t)win * 64 * Cc + (size_t)head * 32;

  if (t < 64) {                       // waves 0,1 (wave-uniform): Q (normalized) + V
    const float* qr = q + base + (size_t)t * Cc;
    float tmp[32], ssum = 0.f;
#pragma unroll
    for (int i = 0; i < 32; ++i) { tmp[i] = qr[i]; ssum += tmp[i] * tmp[i]; }
    float invn = 1.f / fmaxf(sqrtf(ssum), 1e-12f);
#pragma unroll
    for (int i = 0; i < 32; ++i) Qh[t][i] = (_Float16)(tmp[i] * invn);
    const float* vr = v + base + (size_t)t * Cc;
#pragma unroll
    for (int i = 0; i < 32; ++i) Vh[t][i] = (_Float16)vr[i];
  } else {                            // waves 2,3: K (normalized)
    int r = t - 64;
    const float* kr = k + base + (size_t)r * Cc;
    float tmp[32], ssum = 0.f;
#pragma unroll
    for (int i = 0; i < 32; ++i) { tmp[i] = kr[i]; ssum += tmp[i] * tmp[i]; }
    float invn = 1.f / fmaxf(sqrtf(ssum), 1e-12f);
#pragma unroll
    for (int i = 0; i < 32; ++i) Kh[r][i] = (_Float16)(tmp[i] * invn);
  }
  __syncthreads();

  const float ls = expf(fminf(lsc[head], 4.60517018598809f)); // log(100)
  const int lane = t & 31, wv = t >> 5, g = lane >> 4, ln_ = lane & 15;
  const int wh = (win / nW) % nW, ww = win % nW;

  // ---- S = Qn Kn^T * ls + bias (+mask) ----
  for (int tn = 0; tn < 4; ++tn) {
    v16h a, b;
#pragma unroll
    for (int vv = 0; vv < 8; ++vv) {
      int ka = kpatA(vv, g);
      a[2 * vv]     = Qh[wv * 16 + ln_][ka];
      a[2 * vv + 1] = Qh[wv * 16 + ln_][ka + 1];
      int kb = kpatB(vv, g);
      b[2 * vv]     = Kh[tn * 16 + ln_][kb];
      b[2 * vv + 1] = Kh[tn * 16 + ln_][kb + 1];
    }
    v8f c = {};
    c = __builtin_amdgcn_wmma_f32_16x16x32_f16(false, a, false, b, (short)0, c, false, false);
#pragma unroll
    for (int r = 0; r < 8; ++r) {
      int m = wv * 16 + r + 8 * g;
      int n = tn * 16 + ln_;
      float sv = c[r] * ls + biasTab[head * 4096 + m * 64 + n];
      if (sh) {
        int hm = wh * 8 + (m >> 3), wm = ww * 8 + (m & 7);
        int hn = wh * 8 + (n >> 3), wn = ww * 8 + (n & 7);
        int im = ((hm < Hc - 8) ? 0 : ((hm < Hc - 4) ? 1 : 2)) * 3 +
                 ((wm < Hc - 8) ? 0 : ((wm < Hc - 4) ? 1 : 2));
        int in_ = ((hn < Hc - 8) ? 0 : ((hn < Hc - 4) ? 1 : 2)) * 3 +
                  ((wn < Hc - 8) ? 0 : ((wn < Hc - 4) ? 1 : 2));
        if (im != in_) sv -= 100.f;
      }
      S[m][n] = sv;
    }
  }
  __syncthreads();

  // ---- softmax (row per thread, waves 2,3 idle: wave-uniform) ----
  if (t < 64) {
    float mx = -1e30f;
    for (int j = 0; j < 64; ++j) mx = fmaxf(mx, S[t][j]);
    float sum = 0.f;
    for (int j = 0; j < 64; ++j) sum += expf(S[t][j] - mx);
    float inv = 1.f / sum;
    for (int j = 0; j < 64; ++j) P[t][j] = (_Float16)(expf(S[t][j] - mx) * inv);
  }
  __syncthreads();

  // ---- O = P V ----
  for (int tn = 0; tn < 2; ++tn) {
    v8f acc = {};
    for (int kc = 0; kc < 2; ++kc) {
      v16h a, b;
#pragma unroll
      for (int vv = 0; vv < 8; ++vv) {
        int ka = kc * 32 + kpatA(vv, g);
        a[2 * vv]     = P[wv * 16 + ln_][ka];
        a[2 * vv + 1] = P[wv * 16 + ln_][ka + 1];
        int kb = kc * 32 + kpatB(vv, g);
        b[2 * vv]     = Vh[kb][tn * 16 + ln_];
        b[2 * vv + 1] = Vh[kb + 1][tn * 16 + ln_];
      }
      acc = __builtin_amdgcn_wmma_f32_16x16x32_f16(false, a, false, b, (short)0, acc, false, false);
    }
#pragma unroll
    for (int r = 0; r < 8; ++r) {
      int m = wv * 16 + r + 8 * g;
      int n = tn * 16 + ln_;
      o[((size_t)win * 64 + m) * Cc + head * 32 + n] = acc[r];
    }
  }
}

// ============================================================================
// Host orchestration
// ============================================================================
extern "C" void kernel_launch(void* const* d_in, const int* in_sizes, int n_in,
                              void* d_out, int out_size, void* d_ws, size_t ws_size,
                              hipStream_t stream)
{
  (void)in_sizes; (void)n_in; (void)out_size; (void)ws_size;
  const int depths[4]   = {2, 2, 6, 2};
  const int headsArr[4] = {3, 6, 12, 24};

  // --- decode params: JAX pytree order (dict keys sorted, None dropped) ---
  int pi = 0;
  auto In = [&](int i) { return (const float*)d_in[i]; };
  const float* conv_b  = In(pi++);
  const float* conv_w  = In(pi++);
  const float* norm0_b = In(pi++);
  const float* norm0_g = In(pi++);
  // block leaves sorted: bk bo bq bv cpb_b1 cpb_w1 cpb_w2 logit_scale
  //                      mb1 mb2 mw1 mw2 nb1 nb2 ng1 ng2 wk wo wq wv
  const float* blk[4][6][20];
  const float* mrg[3][3]; // b, g, w
  for (int s = 0; s < 4; ++s) {
    for (int b = 0; b < depths[s]; ++b)
      for (int j = 0; j < 20; ++j) blk[s][b][j] = In(pi++);
    if (s < 3)
      for (int j = 0; j < 3; ++j) mrg[s][j] = In(pi++);
  }
  const float* x_in = In(pi++);

  // --- workspace carve ---
  float* Wp = (float*)d_ws;
  size_t off = 0;
  auto carve = [&](size_t n) { float* p = Wp + off; off += n; return p; };
  const size_t TC0 = (size_t)16 * 64 * 64 * 96;
  float* xbuf = carve(TC0);
  float* b1 = carve(TC0);
  float* b2 = carve(TC0);
  float* b3 = carve(TC0);
  float* b4 = carve(TC0);
  float* b5 = carve(TC0);
  float* hid = carve(TC0 * 4);
  float* biasTab = carve((size_t)24 * 64 * 64);

  float* outF = (float*)d_out;

  // --- patch embed + norm0 -> feats[0] ---
  patch_embed_kernel<<<dim3((unsigned)(TC0 / 256)), 256, 0, stream>>>(x_in, conv_w, conv_b, b1);
  ln_kernel<<<dim3((unsigned)(16 * 64 * 64 / 8)), 256, 0, stream>>>(b1, norm0_g, norm0_b, nullptr, xbuf, 96);
  (void)hipMemcpyAsync(outF, xbuf, TC0 * sizeof(float), hipMemcpyDeviceToDevice, stream);
  size_t outOff = TC0;

  int Hc = 64, Cc = 96;
  for (int s = 0; s < 4; ++s) {
    int heads = headsArr[s];
    int nW = Hc / 8;
    int nWin = 16 * nW * nW;
    int toks = 16 * Hc * Hc;
    for (int bi = 0; bi < depths[s]; ++bi) {
      const float* const* P = blk[s][bi];
      int sh = ((bi & 1) && Hc > 8) ? 4 : 0;
      size_t totW = (size_t)toks * Cc;
      unsigned ncpy = (unsigned)((totW + 255) / 256);

      cpb_kernel<<<1, 256, 0, stream>>>(P[5], P[4], P[6], biasTab, heads);
      win_part_kernel<<<ncpy, 256, 0, stream>>>(xbuf, b1, Hc, Cc, sh, nW);

      dim3 gg(Cc / 32, toks / 64);
      gemm_wmma_kernel<0><<<gg, 256, 0, stream>>>(b1, P[18], P[2], b2, Cc, Cc); // Q
      gemm_wmma_kernel<0><<<gg, 256, 0, stream>>>(b1, P[16], P[0], b3, Cc, Cc); // K
      gemm_wmma_kernel<0><<<gg, 256, 0, stream>>>(b1, P[19], P[3], b4, Cc, Cc); // V

      attn_kernel<<<dim3(nWin, heads), 128, 0, stream>>>(b2, b3, b4, P[7], biasTab, b5, Cc, nW, Hc, sh);

      gemm_wmma_kernel<0><<<gg, 256, 0, stream>>>(b5, P[17], P[1], b1, Cc, Cc); // out-proj
      win_rev_kernel<<<ncpy, 256, 0, stream>>>(b1, b2, Hc, Cc, sh, nW);
      // res-post-norm: x = x + LN(attn_out)
      ln_kernel<<<dim3((unsigned)(toks / 8)), 256, 0, stream>>>(b2, P[14], P[12], xbuf, xbuf, Cc);

      // MLP
      dim3 g1(4 * Cc / 32, toks / 64);
      gemm_wmma_kernel<1><<<g1, 256, 0, stream>>>(xbuf, P[10], P[8], hid, 4 * Cc, Cc);
      dim3 g2(Cc / 32, toks / 64);
      gemm_wmma_kernel<0><<<g2, 256, 0, stream>>>(hid, P[11], P[9], b1, Cc, 4 * Cc);
      ln_kernel<<<dim3((unsigned)(toks / 8)), 256, 0, stream>>>(b1, P[15], P[13], xbuf, xbuf, Cc);
    }
    if (s < 3) {
      int H2 = Hc / 2;
      int toks2 = 16 * H2 * H2;
      size_t totM = (size_t)toks2 * 4 * Cc;
      merge_gather_kernel<<<dim3((unsigned)((totM + 255) / 256)), 256, 0, stream>>>(xbuf, b1, Hc, Cc);
      dim3 gm(2 * Cc / 32, toks2 / 64);
      gemm_wmma_kernel<0><<<gm, 256, 0, stream>>>(b1, mrg[s][2], nullptr, b2, 2 * Cc, 4 * Cc);
      ln_kernel<<<dim3((unsigned)(toks2 / 8)), 256, 0, stream>>>(b2, mrg[s][1], mrg[s][0], nullptr, xbuf, 2 * Cc);
      Hc = H2;
      Cc *= 2;
    }
    size_t fs = (size_t)16 * Hc * Hc * Cc;
    (void)hipMemcpyAsync(outF + outOff, xbuf, fs * sizeof(float), hipMemcpyDeviceToDevice, stream);
    outOff += fs;
  }
}
